// GeometricAttention_77945066487961
// MI455X (gfx1250) — matrix operand order repaired and checked by reading, not compiled, not run on hardware
//
#include <hip/hip_runtime.h>

typedef __attribute__((ext_vector_type(16))) _Float16 v16h;
typedef __attribute__((ext_vector_type(8)))  float    v8f;

#define NBATCH 2
#define NSEQ   2048
#define DM     384
#define NH     8
#define HD     48
#define HDP    64
#define QKVN   (3*DM)

// k-index for slot s (0..15) of a 16-bit A operand in this file's contiguous k order (k = 16*hv + s)
__device__ __forceinline__ int a_k(int s, int hv) {
  return hv * 16 + s;
}

__device__ __forceinline__ float rmax16(float v) {
  #pragma unroll
  for (int m = 1; m < 16; m <<= 1) v = fmaxf(v, __shfl_xor(v, m, 32));
  return v;
}
__device__ __forceinline__ float rsum16(float v) {
  #pragma unroll
  for (int m = 1; m < 16; m <<= 1) v += __shfl_xor(v, m, 32);
  return v;
}

__global__ void geoattn_zero_kernel(unsigned int* __restrict__ p, int n) {
  int idx = blockIdx.x * blockDim.x + threadIdx.x;
  int stride = gridDim.x * blockDim.x;
  for (int i = idx; i < n; i += stride) p[i] = 0u;
}

// ---------------- QKV projection:  qkv = x @ W_qkv + b_qkv  -> f16 Q/K (pad 64), V (48)
__global__ void geoattn_qkv_wmma(const float* __restrict__ x,
                                 const float* __restrict__ Wqkv,
                                 const float* __restrict__ bqkv,
                                 _Float16* __restrict__ Qp,
                                 _Float16* __restrict__ Kp,
                                 _Float16* __restrict__ Vp) {
  const int lane = threadIdx.x & 31;
  const int hv = lane >> 4;
  const int ln = lane & 15;
  const int row0 = blockIdx.y * 16;   // flattened (b*n) rows
  const int col0 = blockIdx.x * 16;   // 0..1151

  v8f acc = {};
  for (int k0 = 0; k0 < DM; k0 += 32) {
    v16h a, bm;
    #pragma unroll
    for (int s = 0; s < 16; ++s) {
      a[s]  = (_Float16)x[(size_t)(row0 + ln) * DM + (k0 + a_k(s, hv))];
      bm[s] = (_Float16)Wqkv[(size_t)(k0 + hv * 16 + s) * QKVN + col0 + ln];
    }
    acc = __builtin_amdgcn_wmma_f32_16x16x32_f16(false, a, false, bm,
                                                 (short)0, acc, false, false);
  }

  const float bias = bqkv[col0 + ln];
  const int sec = col0 / DM;          // 0=Q, 1=K, 2=V (tiles never straddle)
  const int cc0 = col0 % DM;
  const int hh  = cc0 / HD;
  const int dd  = (cc0 % HD) + ln;    // 48 = 3*16, tile stays inside one head
  #pragma unroll
  for (int r = 0; r < 8; ++r) {
    int grow = row0 + hv * 8 + r;
    int bb = grow / NSEQ, ii = grow % NSEQ;
    float v = acc[r] + bias;
    size_t base = ((size_t)(bb * NH + hh)) * NSEQ + ii;
    if (sec == 0)      Qp[base * HDP + dd] = (_Float16)v;
    else if (sec == 1) Kp[base * HDP + dd] = (_Float16)v;
    else               Vp[base * HD  + dd] = (_Float16)v;
  }
}

// ---------------- Fused flash attention with distance bias -----------------
__global__ void geoattn_flash_wmma(const _Float16* __restrict__ Qp,
                                   const _Float16* __restrict__ Kp,
                                   const _Float16* __restrict__ Vp,
                                   const float* __restrict__ coords,
                                   const float* __restrict__ Wdist,
                                   const float* __restrict__ bdist,
                                   _Float16* __restrict__ AO) {
  __shared__ _Float16 smem[16 * 32];
  const int lane = threadIdx.x & 31;
  const int hv = lane >> 4;
  const int ln = lane & 15;
  const int i0 = blockIdx.x * 16;
  const int hh = blockIdx.y;
  const int bb = blockIdx.z;
  const size_t baseR = ((size_t)(bb * NH + hh)) * NSEQ;

  // Q tile (16 rows x 64 padded d), pre-scaled by 1/sqrt(48)
  const float sc = 0.14433756729740643f;
  v16h qa0, qa1;
  #pragma unroll
  for (int s = 0; s < 16; ++s) {
    int k = a_k(s, hv);
    qa0[s] = (_Float16)((float)Qp[(baseR + i0 + ln) * HDP + k] * sc);
    qa1[s] = (_Float16)((float)Qp[(baseR + i0 + ln) * HDP + 32 + k] * sc);
  }

  float ci[8][3];
  #pragma unroll
  for (int r = 0; r < 8; ++r) {
    size_t ir = (size_t)bb * NSEQ + (i0 + hv * 8 + r);
    ci[r][0] = coords[ir * 3 + 0];
    ci[r][1] = coords[ir * 3 + 1];
    ci[r][2] = coords[ir * 3 + 2];
  }
  const float wD = Wdist[hh];
  const float bD = bdist[hh];

  float mrow[8], lrow[8];
  #pragma unroll
  for (int r = 0; r < 8; ++r) { mrow[r] = -3.0e38f; lrow[r] = 0.0f; }
  v8f o0 = {}, o1 = {}, o2 = {};

  for (int j0 = 0; j0 < NSEQ; j0 += 32) {
    v8f st[2];
    #pragma unroll
    for (int jt = 0; jt < 2; ++jt) {
      const int jb = j0 + jt * 16;
      v16h kb0, kb1;
      #pragma unroll
      for (int s = 0; s < 16; ++s) {
        int kk = hv * 16 + s;            // B layout: n = ln, k = kk
        kb0[s] = Kp[(baseR + jb + ln) * HDP + kk];
        kb1[s] = Kp[(baseR + jb + ln) * HDP + 32 + kk];
      }
      v8f sa = {};
      sa = __builtin_amdgcn_wmma_f32_16x16x32_f16(false, qa0, false, kb0,
                                                  (short)0, sa, false, false);
      sa = __builtin_amdgcn_wmma_f32_16x16x32_f16(false, qa1, false, kb1,
                                                  (short)0, sa, false, false);
      // distance bias: dist(i, j) * w_h + b_h
      size_t jr = (size_t)bb * NSEQ + (jb + ln);
      float cj0 = coords[jr * 3 + 0], cj1 = coords[jr * 3 + 1], cj2 = coords[jr * 3 + 2];
      #pragma unroll
      for (int r = 0; r < 8; ++r) {
        float d0 = ci[r][0] - cj0, d1 = ci[r][1] - cj1, d2 = ci[r][2] - cj2;
        float dist = sqrtf(d0 * d0 + d1 * d1 + d2 * d2);
        sa[r] += dist * wD + bD;
      }
      st[jt] = sa;
    }

    // online softmax (rows live in lane halves; xor<=8 stays in-half)
    #pragma unroll
    for (int r = 0; r < 8; ++r) {
      float mx   = rmax16(fmaxf(st[0][r], st[1][r]));
      float mnew = fmaxf(mrow[r], mx);
      float alpha = __expf(mrow[r] - mnew);
      float p0 = __expf(st[0][r] - mnew);
      float p1 = __expf(st[1][r] - mnew);
      lrow[r] = lrow[r] * alpha + rsum16(p0 + p1);
      mrow[r] = mnew;
      o0[r] *= alpha; o1[r] *= alpha; o2[r] *= alpha;
      smem[(hv * 8 + r) * 32 + ln]      = (_Float16)p0;
      smem[(hv * 8 + r) * 32 + 16 + ln] = (_Float16)p1;
    }
    __syncthreads();
    v16h pa;
    #pragma unroll
    for (int s = 0; s < 16; ++s) pa[s] = smem[ln * 32 + a_k(s, hv)];
    __syncthreads();

    // O += P (16x32) @ V (32x48) in three 16-wide d-chunks
    v16h vb;
    #pragma unroll
    for (int s = 0; s < 16; ++s)
      vb[s] = Vp[(baseR + j0 + hv * 16 + s) * HD + 0 + ln];
    o0 = __builtin_amdgcn_wmma_f32_16x16x32_f16(false, pa, false, vb, (short)0, o0, false, false);
    #pragma unroll
    for (int s = 0; s < 16; ++s)
      vb[s] = Vp[(baseR + j0 + hv * 16 + s) * HD + 16 + ln];
    o1 = __builtin_amdgcn_wmma_f32_16x16x32_f16(false, pa, false, vb, (short)0, o1, false, false);
    #pragma unroll
    for (int s = 0; s < 16; ++s)
      vb[s] = Vp[(baseR + j0 + hv * 16 + s) * HD + 32 + ln];
    o2 = __builtin_amdgcn_wmma_f32_16x16x32_f16(false, pa, false, vb, (short)0, o2, false, false);
  }

  #pragma unroll
  for (int r = 0; r < 8; ++r) {
    int irow = i0 + hv * 8 + r;
    float inv = 1.0f / lrow[r];
    size_t ob = ((size_t)bb * NSEQ + irow) * DM + hh * HD;
    AO[ob + 0  + ln] = (_Float16)(o0[r] * inv);
    AO[ob + 16 + ln] = (_Float16)(o1[r] * inv);
    AO[ob + 32 + ln] = (_Float16)(o2[r] * inv);
  }
}

// ---------------- Output projection: out = AO @ W_out + b_out --------------
__global__ void geoattn_oproj_wmma(const _Float16* __restrict__ AO,
                                   const float* __restrict__ Wout,
                                   const float* __restrict__ bout,
                                   float* __restrict__ out) {
  const int lane = threadIdx.x & 31;
  const int hv = lane >> 4;
  const int ln = lane & 15;
  const int row0 = blockIdx.y * 16;
  const int col0 = blockIdx.x * 16;

  v8f acc = {};
  for (int k0 = 0; k0 < DM; k0 += 32) {
    v16h a, bm;
    #pragma unroll
    for (int s = 0; s < 16; ++s) {
      a[s]  = AO[(size_t)(row0 + ln) * DM + (k0 + a_k(s, hv))];
      bm[s] = (_Float16)Wout[(size_t)(k0 + hv * 16 + s) * DM + col0 + ln];
    }
    acc = __builtin_amdgcn_wmma_f32_16x16x32_f16(false, a, false, bm,
                                                 (short)0, acc, false, false);
  }
  const float bias = bout[col0 + ln];
  #pragma unroll
  for (int r = 0; r < 8; ++r)
    out[(size_t)(row0 + hv * 8 + r) * DM + col0 + ln] = acc[r] + bias;
}

extern "C" void kernel_launch(void* const* d_in, const int* in_sizes, int n_in,
                              void* d_out, int out_size, void* d_ws, size_t ws_size,
                              hipStream_t stream) {
  (void)in_sizes; (void)n_in; (void)out_size; (void)ws_size;
  const float* x      = (const float*)d_in[0];
  const float* coords = (const float*)d_in[1];
  const float* Wqkv   = (const float*)d_in[2];
  const float* bqkv   = (const float*)d_in[3];
  const float* Wdist  = (const float*)d_in[4];
  const float* bdist  = (const float*)d_in[5];
  const float* Wout   = (const float*)d_in[6];
  const float* bout   = (const float*)d_in[7];
  float* out = (float*)d_out;

  char* ws = (char*)d_ws;
  _Float16* Qp = (_Float16*)(ws + 0);                      // 2*8*2048*64 f16 = 4 MB
  _Float16* Kp = (_Float16*)(ws + 4u * 1024 * 1024);       // 4 MB
  _Float16* Vp = (_Float16*)(ws + 8u * 1024 * 1024);       // 2*8*2048*48 f16 = 3 MB
  _Float16* AO = (_Float16*)(ws + 11u * 1024 * 1024);      // 2*2048*384 f16 = 3 MB

  // zero Q/K pad regions (whole padded arrays) every call: 8 MB = 2M dwords
  geoattn_zero_kernel<<<2048, 256, 0, stream>>>((unsigned int*)ws, 2 * 1024 * 1024);

  // QKV: rows = b*n = 4096 -> 256 tiles; cols = 1152 -> 72 tiles
  geoattn_qkv_wmma<<<dim3(QKVN / 16, (NBATCH * NSEQ) / 16), 32, 0, stream>>>(
      x, Wqkv, bqkv, Qp, Kp, Vp);

  // fused attention: (n/16, heads, batch), one wave per block
  geoattn_flash_wmma<<<dim3(NSEQ / 16, NH, NBATCH), 32, 0, stream>>>(
      Qp, Kp, Vp, coords, Wdist, bdist, AO);

  // output projection: rows 4096 -> 256 tiles; cols 384 -> 24 tiles
  geoattn_oproj_wmma<<<dim3(DM / 16, (NBATCH * NSEQ) / 16), 32, 0, stream>>>(
      AO, Wout, bout, out);
}
